// SimpleGraphAEModel_49246095016173
// MI455X (gfx1250) — compile-verified, matrix operand order checked by reading, not run on hardware
//
#include <hip/hip_runtime.h>

typedef __attribute__((ext_vector_type(16))) _Float16 v16h;
typedef __attribute__((ext_vector_type(8)))  _Float16 v8h;
typedef __attribute__((ext_vector_type(8)))  float    v8f;

#define LDF 40  // LDS row stride in halves: 80B = 20 banks -> conflict-free b128 reads

// ---------------- utility kernels ----------------

__global__ void zero_f32_k(float* __restrict__ p, int n) {
  int i = blockIdx.x * blockDim.x + threadIdx.x;
  if (i < n) p[i] = 0.0f;
}

// f32 [R,C] -> f16 [Rp,Cp] zero-padded (row-major), for activations (A matrices)
__global__ void cvt_pad_f16_k(const float* __restrict__ src, _Float16* __restrict__ dst,
                              int R, int C, int Cp, int total) {
  int i = blockIdx.x * blockDim.x + threadIdx.x;
  if (i >= total) return;
  int r = i / Cp;
  int c = i - r * Cp;
  float v = 0.0f;
  if (r < R && c < C) v = src[(size_t)r * C + c];
  dst[i] = (_Float16)v;
}

// f32 W[K,Nact] -> f16 Bt[Np,Kp] TRANSPOSED + zero-padded (weights, B matrices)
__global__ void cvt_pad_f16_t_k(const float* __restrict__ src, _Float16* __restrict__ dst,
                                int K, int Nact, int Kp, int total) {
  int i = blockIdx.x * blockDim.x + threadIdx.x;
  if (i >= total) return;
  int n = i / Kp;
  int k = i - n * Kp;
  float v = 0.0f;
  if (k < K && n < Nact) v = src[(size_t)k * Nact + n];
  dst[i] = (_Float16)v;
}

// ---------------- WMMA GEMM ----------------
// C[M,Nact] = A_f16[M,Kp] * Bt_f16[Np,Kp]^T + bias   (Bt is N-major, K-contiguous)
// optional f32 and/or f16 outputs

__global__ __launch_bounds__(256)
void gemm_wmma_f16_k(const _Float16* __restrict__ A, const _Float16* __restrict__ Bt,
                     const float* __restrict__ bias,
                     float* __restrict__ C32, _Float16* __restrict__ C16,
                     int M, int Kp, int Np, int Nact) {
  __shared__ _Float16 Alds[128 * LDF];
  __shared__ _Float16 Blds[64 * LDF];

  const int tid  = threadIdx.x;
  const int lane = tid & 31;
  const int wave = tid >> 5;
  const int rowBase = blockIdx.y * 128;
  const int bn0     = blockIdx.x * 64;
  const int wr = wave & 3;          // 4 row groups
  const int wc = wave >> 2;         // 2 col groups
  const int row0 = wr * 32;
  const int col0 = wc * 32;
  const int m16   = lane & 15;
  const int chalf = lane >> 4;      // lane half selects K +0/+8 (and +16/+24)

  // branch-free A row indices (clamp; garbage rows are masked at the store)
  const int arow0 = tid >> 1;                       // chunks 0..255 -> rows 0..127? (2 chunks/thread)
  v8f acc[2][2] = {};

  for (int kb = 0; kb < Kp; kb += 32) {
    // A tile: 128 rows x 32 halves = 512 v8h chunks, 2 per thread
#pragma unroll
    for (int i = 0; i < 2; ++i) {
      int chunk = tid + i * 256;
      int row = chunk >> 2;
      int kc  = chunk & 3;
      int gr  = rowBase + row;
      gr = gr < M ? gr : (M - 1);
      v8h v = *(const v8h*)(A + (size_t)gr * Kp + kb + kc * 8);
      *(v8h*)&Alds[row * LDF + kc * 8] = v;
    }
    // B tile: 64 rows (n) x 32 halves = 256 v8h chunks, 1 per thread (Np padded)
    {
      int row = tid >> 2;
      int kc  = tid & 3;
      v8h v = *(const v8h*)(Bt + (size_t)(bn0 + row) * Kp + kb + kc * 8);
      *(v8h*)&Blds[row * LDF + kc * 8] = v;
    }
    __syncthreads();

    v16h af[2], bf[2];
#pragma unroll
    for (int s = 0; s < 2; ++s) {
      const _Float16* ap = &Alds[(row0 + s * 16 + m16) * LDF + 8 * chalf];
      v8h lo = *(const v8h*)ap;          // K = 8c + 0..7
      v8h hi = *(const v8h*)(ap + 16);   // K = 16 + 8c + 0..7
      af[s] = __builtin_shufflevector(lo, hi, 0,1,2,3,4,5,6,7,8,9,10,11,12,13,14,15);
    }
#pragma unroll
    for (int t = 0; t < 2; ++t) {
      const _Float16* bp = &Blds[(col0 + t * 16 + m16) * LDF + 8 * chalf];
      v8h lo = *(const v8h*)bp;
      v8h hi = *(const v8h*)(bp + 16);
      bf[t] = __builtin_shufflevector(lo, hi, 0,1,2,3,4,5,6,7,8,9,10,11,12,13,14,15);
    }
#pragma unroll
    for (int s = 0; s < 2; ++s)
#pragma unroll
      for (int t = 0; t < 2; ++t)
        acc[s][t] = __builtin_amdgcn_wmma_f32_16x16x32_f16(
            false, af[s], false, bf[t], (short)0, acc[s][t], false, false);
    __syncthreads();
  }

  // C/D layout: VGPR r, lanes 0-15: (M=r, N=lane); lanes 16-31: (M=r+8, N=lane-16)
#pragma unroll
  for (int s = 0; s < 2; ++s)
#pragma unroll
    for (int t = 0; t < 2; ++t)
#pragma unroll
      for (int r = 0; r < 8; ++r) {
        int gr = rowBase + row0 + s * 16 + r + 8 * chalf;
        int gc = bn0 + col0 + t * 16 + m16;
        if (gr < M && gc < Nact) {
          float v = acc[s][t][r] + bias[gc];
          if (C32) C32[(size_t)gr * Nact + gc] = v;
          if (C16) C16[(size_t)gr * Nact + gc] = (_Float16)v;
        }
      }
  (void)arow0;
}

// ---------------- edge softmax kernels ----------------

__device__ __forceinline__ unsigned encf(float x) {
  unsigned u = __float_as_uint(x);
  return (u & 0x80000000u) ? ~u : (u | 0x80000000u);
}
__device__ __forceinline__ float decf(unsigned u) {
  return __uint_as_float((u & 0x80000000u) ? (u ^ 0x80000000u) : ~u);
}
__device__ __forceinline__ void edge_sd(const int* __restrict__ ei, int E, int e,
                                        int& s, int& d) {
  if (e < E) { s = ei[e]; d = ei[E + e]; }
  else       { s = d = e - E; }           // self-loops appended
}

// wave per edge: logit = dot(leakyrelu(xl[src]+xr[dst]), att); segment max
__global__ __launch_bounds__(256)
void edge_logits_k(const float* __restrict__ xl, const float* __restrict__ xr,
                   const float* __restrict__ att, const int* __restrict__ ei,
                   int E, int Etot, int dout,
                   float* __restrict__ logits, unsigned* __restrict__ menc) {
  int e = blockIdx.x * 8 + (threadIdx.x >> 5);
  if (e >= Etot) return;
  int lane = threadIdx.x & 31;
  int s, d; edge_sd(ei, E, e, s, d);
  const float* pl = xl + (size_t)s * dout;
  const float* pr = xr + (size_t)d * dout;
  float acc = 0.0f;
  for (int k = lane; k < dout; k += 32) {
    float h = pl[k] + pr[k];
    h = h > 0.0f ? h : 0.2f * h;
    acc += h * att[k];
  }
  for (int off = 16; off > 0; off >>= 1) acc += __shfl_down(acc, off);
  if (lane == 0) {
    logits[e] = acc;
    atomicMax(&menc[d], encf(acc));
  }
}

__global__ void edge_exp_k(const float* __restrict__ logits, const unsigned* __restrict__ menc,
                           const int* __restrict__ ei, int E, int Etot,
                           float* __restrict__ ex, float* __restrict__ ssum) {
  int e = blockIdx.x * blockDim.x + threadIdx.x;
  if (e >= Etot) return;
  int s, d; edge_sd(ei, E, e, s, d);
  float v = __expf(logits[e] - decf(menc[d]));
  ex[e] = v;
  atomicAdd(&ssum[d], v);
}

// wave per edge: out[dst] += alpha * xl[src]
__global__ __launch_bounds__(256)
void edge_scatter_k(const float* __restrict__ xl, const float* __restrict__ ex,
                    const float* __restrict__ ssum, const int* __restrict__ ei,
                    int E, int Etot, int dout, float* __restrict__ out) {
  int e = blockIdx.x * 8 + (threadIdx.x >> 5);
  if (e >= Etot) return;
  int lane = threadIdx.x & 31;
  int s, d; edge_sd(ei, E, e, s, d);
  float alpha = ex[e] / ssum[d];
  const float* pl = xl + (size_t)s * dout;
  float* po = out + (size_t)d * dout;
  for (int k = lane; k < dout; k += 32)
    atomicAdd(&po[k], alpha * pl[k]);
}

// gat output + bias -> optional f32 out and f16 copy for next GEMM
__global__ void post_bias_k(const float* __restrict__ gat, const float* __restrict__ bias,
                            float* __restrict__ out32, _Float16* __restrict__ out16,
                            int dout, int total) {
  int i = blockIdx.x * blockDim.x + threadIdx.x;
  if (i >= total) return;
  float v = gat[i] + bias[i % dout];
  if (out32) out32[i] = v;
  out16[i] = (_Float16)v;
}

// ---------------- host launcher ----------------

extern "C" void kernel_launch(void* const* d_in, const int* in_sizes, int n_in,
                              void* d_out, int out_size, void* d_ws, size_t ws_size,
                              hipStream_t stream) {
  (void)n_in; (void)out_size; (void)ws_size;
  const int N = 10000, DIN = 1000, D0 = 256, D1 = 128;
  const int DINP = 1024;  // padded 1000 -> 1024 (K for proj_in, N for proj_out)

  const float* x     = (const float*)d_in[0];
  const int*   ei    = (const int*)  d_in[1];
  const float* Wpi   = (const float*)d_in[2];
  const float* bpi   = (const float*)d_in[3];
  const float* Wl1   = (const float*)d_in[4];
  const float* bl1   = (const float*)d_in[5];
  const float* Wr1   = (const float*)d_in[6];
  const float* br1   = (const float*)d_in[7];
  const float* att1  = (const float*)d_in[8];
  const float* bias1 = (const float*)d_in[9];
  const float* Wl2   = (const float*)d_in[10];
  const float* bl2   = (const float*)d_in[11];
  const float* Wr2   = (const float*)d_in[12];
  const float* br2   = (const float*)d_in[13];
  const float* att2  = (const float*)d_in[14];
  const float* bias2 = (const float*)d_in[15];
  const float* Wpo   = (const float*)d_in[16];
  const float* bpo   = (const float*)d_in[17];

  const int E    = in_sizes[1] / 2;
  const int Etot = E + N;

  // workspace layout
  char* ws = (char*)d_ws;
  size_t o = 0;
  auto al = [&](size_t b) -> char* {
    char* p = ws + o; o = (o + b + 255) & ~(size_t)255; return p;
  };
  char*     buf0 = al((size_t)N * DINP * 2);             // x16; later xl2|xr2 (f32)
  _Float16* w16  = (_Float16*)al((size_t)DINP * D0 * 2); // transposed weight staging (512KB)
  _Float16* h16  = (_Float16*)al((size_t)N * D0 * 2);
  float*    xl1  = (float*)   al((size_t)N * D1 * 4);
  float*    xr1  = (float*)   al((size_t)N * D1 * 4);
  float*    gat  = (float*)   al((size_t)N * D0 * 4);    // reused for both layers
  _Float16* z16  = (_Float16*)al((size_t)N * D1 * 2);
  _Float16* d16  = (_Float16*)al((size_t)N * D0 * 2);
  float*    lg   = (float*)   al((size_t)Etot * 4);
  float*    exb  = (float*)   al((size_t)Etot * 4);
  unsigned* menc = (unsigned*)al((size_t)N * 4);
  float*    ssum = (float*)   al((size_t)N * 4);

  _Float16* x16 = (_Float16*)buf0;
  float*    xl2 = (float*)buf0;
  float*    xr2 = (float*)(buf0 + (size_t)N * D0 * 4);

  float* out  = (float*)d_out;             // x_hat: N*1000
  float* zout = out + (size_t)N * DIN;     // z: N*128

  const int T = 256;
  auto blk = [](long long n, int t) { return (unsigned)((n + t - 1) / t); };
  dim3 eblk(blk(Etot, 8));                 // wave-per-edge kernels

  // ---- proj_in: h16 = f16(x @ Wpi + bpi) ----
  hipLaunchKernelGGL(cvt_pad_f16_k, dim3(blk((long long)N * DINP, T)), dim3(T), 0, stream,
                     x, x16, N, DIN, DINP, N * DINP);
  hipLaunchKernelGGL(cvt_pad_f16_t_k, dim3(blk(D0 * DINP, T)), dim3(T), 0, stream,
                     Wpi, w16, DIN, D0, DINP, D0 * DINP);           // Bt [256,1024]
  hipLaunchKernelGGL(gemm_wmma_f16_k, dim3(D0 / 64, blk(N, 128)), dim3(T), 0, stream,
                     x16, w16, bpi, (float*)nullptr, h16, N, DINP, D0, D0);

  // ---- GAT layer 1 (256 -> 128) ----
  hipLaunchKernelGGL(cvt_pad_f16_t_k, dim3(blk(D1 * D0, T)), dim3(T), 0, stream,
                     Wl1, w16, D0, D1, D0, D1 * D0);                // Bt [128,256]
  hipLaunchKernelGGL(gemm_wmma_f16_k, dim3(D1 / 64, blk(N, 128)), dim3(T), 0, stream,
                     h16, w16, bl1, xl1, (_Float16*)nullptr, N, D0, D1, D1);
  hipLaunchKernelGGL(cvt_pad_f16_t_k, dim3(blk(D1 * D0, T)), dim3(T), 0, stream,
                     Wr1, w16, D0, D1, D0, D1 * D0);
  hipLaunchKernelGGL(gemm_wmma_f16_k, dim3(D1 / 64, blk(N, 128)), dim3(T), 0, stream,
                     h16, w16, br1, xr1, (_Float16*)nullptr, N, D0, D1, D1);

  hipLaunchKernelGGL(zero_f32_k, dim3(blk(N, T)), dim3(T), 0, stream, (float*)menc, N);
  hipLaunchKernelGGL(zero_f32_k, dim3(blk(N, T)), dim3(T), 0, stream, ssum, N);
  hipLaunchKernelGGL(zero_f32_k, dim3(blk((long long)N * D1, T)), dim3(T), 0, stream, gat, N * D1);
  hipLaunchKernelGGL(edge_logits_k, eblk, dim3(T), 0, stream,
                     xl1, xr1, att1, ei, E, Etot, D1, lg, menc);
  hipLaunchKernelGGL(edge_exp_k, dim3(blk(Etot, T)), dim3(T), 0, stream,
                     lg, menc, ei, E, Etot, exb, ssum);
  hipLaunchKernelGGL(edge_scatter_k, eblk, dim3(T), 0, stream,
                     xl1, exb, ssum, ei, E, Etot, D1, gat);
  hipLaunchKernelGGL(post_bias_k, dim3(blk((long long)N * D1, T)), dim3(T), 0, stream,
                     gat, bias1, zout, z16, D1, N * D1);

  // ---- GAT layer 2 (128 -> 256) ----
  hipLaunchKernelGGL(cvt_pad_f16_t_k, dim3(blk(D0 * D1, T)), dim3(T), 0, stream,
                     Wl2, w16, D1, D0, D1, D0 * D1);                // Bt [256,128]
  hipLaunchKernelGGL(gemm_wmma_f16_k, dim3(D0 / 64, blk(N, 128)), dim3(T), 0, stream,
                     z16, w16, bl2, xl2, (_Float16*)nullptr, N, D1, D0, D0);
  hipLaunchKernelGGL(cvt_pad_f16_t_k, dim3(blk(D0 * D1, T)), dim3(T), 0, stream,
                     Wr2, w16, D1, D0, D1, D0 * D1);
  hipLaunchKernelGGL(gemm_wmma_f16_k, dim3(D0 / 64, blk(N, 128)), dim3(T), 0, stream,
                     z16, w16, br2, xr2, (_Float16*)nullptr, N, D1, D0, D0);

  hipLaunchKernelGGL(zero_f32_k, dim3(blk(N, T)), dim3(T), 0, stream, (float*)menc, N);
  hipLaunchKernelGGL(zero_f32_k, dim3(blk(N, T)), dim3(T), 0, stream, ssum, N);
  hipLaunchKernelGGL(zero_f32_k, dim3(blk((long long)N * D0, T)), dim3(T), 0, stream, gat, N * D0);
  hipLaunchKernelGGL(edge_logits_k, eblk, dim3(T), 0, stream,
                     xl2, xr2, att2, ei, E, Etot, D0, lg, menc);
  hipLaunchKernelGGL(edge_exp_k, dim3(blk(Etot, T)), dim3(T), 0, stream,
                     lg, menc, ei, E, Etot, exb, ssum);
  hipLaunchKernelGGL(edge_scatter_k, eblk, dim3(T), 0, stream,
                     xl2, exb, ssum, ei, E, Etot, D0, gat);
  hipLaunchKernelGGL(post_bias_k, dim3(blk((long long)N * D0, T)), dim3(T), 0, stream,
                     gat, bias2, (float*)nullptr, d16, D0, N * D0);

  // ---- proj_out: x_hat = d @ Wpo + bpo ----
  hipLaunchKernelGGL(cvt_pad_f16_t_k, dim3(blk((long long)DINP * D0, T)), dim3(T), 0, stream,
                     Wpo, w16, D0, DIN, D0, DINP * D0);             // Bt [1024,256]
  hipLaunchKernelGGL(gemm_wmma_f16_k, dim3(DINP / 64, blk(N, 128)), dim3(T), 0, stream,
                     d16, w16, bpo, out, (_Float16*)nullptr, N, D0, DINP, DIN);
}
